// WindowAttention_34024730919193
// MI455X (gfx1250) — matrix operand order repaired and checked by reading, not compiled, not run on hardware
//
#include <hip/hip_runtime.h>
#include <hip/hip_bf16.h>

typedef __bf16 bf16_t;
typedef __attribute__((ext_vector_type(16))) __bf16 v16bf;
typedef __attribute__((ext_vector_type(8)))  __bf16 v8bf;
typedef __attribute__((ext_vector_type(8)))  float   v8f;

#define DIM    256
#define HEADS  8
#define NTOK   49
#define NPAD   64
#define HD     32
#define NW     64
#define QKV_ELEMS (3*DIM*DIM)   // 768*256
#define PRJ_ELEMS (DIM*DIM)     // 256*256

// ---------------- WMMA helpers (gfx1250, wave32) ----------------

__device__ __forceinline__ v8f wmma_bf16(v16bf a, v16bf b, v8f c) {
  // v_wmma_f32_16x16x32_bf16 : D = A(16x32) * B(32x16) + C(16x16 f32)
  return __builtin_amdgcn_wmma_f32_16x16x32_bf16(
      false, a, false, b, (short)0, c, false, false);
}

// A-matrix fragment (16x32 bf16). Per ISA 7.12.2: lane = row (mod 16);
// lanes 0-15 hold K 0..7 & 16..23, lanes 16-31 hold K 8..15 & 24..31.
// Row-major source with `stride` elements per row -> two 16B loads.
__device__ __forceinline__ v16bf ldA(const bf16_t* base, int stride,
                                     int row, int kbase, int lane) {
  int r    = lane & 15;
  int koff = (lane >> 4) << 3;
  const bf16_t* p = base + (size_t)(row + r) * stride + kbase + koff;
  v8bf lo = *(const v8bf*)p;
  v8bf hi = *(const v8bf*)(p + 16);
  return __builtin_shufflevector(lo, hi, 0,1,2,3,4,5,6,7,8,9,10,11,12,13,14,15);
}

// B-matrix fragment (32x16 bf16). Same K order as ldA:
// lane = column (mod 16); lanes 0-15 hold K 0..7 & 16..23, lanes 16-31 hold K 8..15 & 24..31.
// Source laid out [col][k] row-major -> two 16B loads.
__device__ __forceinline__ v16bf ldB(const bf16_t* base, int stride,
                                     int col, int kbase, int lane) {
  int c  = col + (lane & 15);
  int kb = kbase + ((lane >> 4) << 3);
  const bf16_t* p = base + (size_t)c * stride + kb;
  v8bf lo = *(const v8bf*)p;
  v8bf hi = *(const v8bf*)(p + 16);
  return __builtin_shufflevector(lo, hi, 0,1,2,3,4,5,6,7,8,9,10,11,12,13,14,15);
}

// ---------------- weight conversion: fp32 -> bf16 in workspace ----------------

__global__ void convert_weights_kernel(const float* __restrict__ qkv_w,
                                       const float* __restrict__ proj_w,
                                       bf16_t* __restrict__ wq,
                                       bf16_t* __restrict__ wp) {
  int i = blockIdx.x * 256 + threadIdx.x;
  if (i < QKV_ELEMS) wq[i] = (bf16_t)qkv_w[i];
  if (i < PRJ_ELEMS) wp[i] = (bf16_t)proj_w[i];
}

// ---------------- fused window attention: one block per window ----------------

__global__ __launch_bounds__(256)
void win_attn_kernel(const float* __restrict__ x,
                     const float* __restrict__ mask,
                     const float* __restrict__ qkv_b,
                     const float* __restrict__ proj_b,
                     const float* __restrict__ bias_table,
                     const int*   __restrict__ rel_index,
                     const bf16_t* __restrict__ wqkv,   // [768][256] bf16
                     const bf16_t* __restrict__ wproj,  // [256][256] bf16
                     float* __restrict__ out) {
  extern __shared__ char smem[];
  bf16_t* sX  = (bf16_t*)(smem);            // [64][256] x tile; later O tile
  bf16_t* sQ  = (bf16_t*)(smem + 32768);    // [8][64][32]  (pre-scaled)
  bf16_t* sK  = (bf16_t*)(smem + 65536);    // [8][64][32]
  bf16_t* sVt = (bf16_t*)(smem + 98304);    // [8][32][64]  (V transposed)
  float*  sB  = (float*) (smem + 131072);   // [8][49*49]   bias+mask
  bf16_t* sP  = (bf16_t*)(smem + 207936);   // [8][64][64]  softmax probs

  const int b    = blockIdx.x;
  const int tid  = threadIdx.x;
  const int lane = tid & 31;
  const int w    = tid >> 5;          // wave id == head id
  const int half = lane >> 4;
  const int lr   = lane & 15;

  // ---- Phase 1: stage x (bf16, zero-padded rows 49..63) + bias/mask table ----
  {
    const float* xb = x + (size_t)b * NTOK * DIM;
    for (int idx = tid; idx < NPAD * (DIM / 4); idx += 256) {
      int row = idx >> 6;
      int c4  = (idx & 63) << 2;
      float4 vv = make_float4(0.f, 0.f, 0.f, 0.f);
      if (row < NTOK) vv = *(const float4*)(xb + row * DIM + c4);
      bf16_t* d = sX + row * DIM + c4;
      d[0] = (bf16_t)vv.x; d[1] = (bf16_t)vv.y;
      d[2] = (bf16_t)vv.z; d[3] = (bf16_t)vv.w;
    }
    const float* mb = mask + (size_t)(b & (NW - 1)) * NTOK * NTOK;
    for (int idx = tid; idx < HEADS * NTOK * NTOK; idx += 256) {
      int h = idx / (NTOK * NTOK);
      int r = idx - h * NTOK * NTOK;
      int ri = rel_index[r];
      sB[idx] = bias_table[ri * HEADS + h] + mb[r];
    }
  }
  __syncthreads();

  // ---- Phase 2: QKV for head w : [64x256] @ [256x96] slices via WMMA ----
  const float scale = 0.17677669529663687f;  // 32^-0.5
  for (int t = 0; t < 3; ++t) {              // 0:Q 1:K 2:V
    int ob = t * DIM + w * HD;               // output column base in [0,768)
    for (int mt = 0; mt < 4; ++mt) {
      for (int nt = 0; nt < 2; ++nt) {
        v8f acc = {};
        #pragma unroll
        for (int kt = 0; kt < 8; ++kt) {
          v16bf a  = ldA(sX, DIM, mt * 16, kt * 32, lane);
          v16bf bb = ldB(wqkv, DIM, ob + nt * 16, kt * 32, lane);
          acc = wmma_bf16(a, bb, acc);
        }
        int   col  = nt * 16 + lr;           // column within head slice [0,32)
        float bias = qkv_b[ob + col];
        int   rowb = mt * 16 + half * 8;
        if (t == 0) {
          bf16_t* d = sQ + ((size_t)w * NPAD + rowb) * HD + col;
          #pragma unroll
          for (int i = 0; i < 8; ++i) d[i * HD] = (bf16_t)((acc[i] + bias) * scale);
        } else if (t == 1) {
          bf16_t* d = sK + ((size_t)w * NPAD + rowb) * HD + col;
          #pragma unroll
          for (int i = 0; i < 8; ++i) d[i * HD] = (bf16_t)(acc[i] + bias);
        } else {                             // V stored transposed: [col][row]
          bf16_t* d = sVt + ((size_t)w * HD + col) * NPAD + rowb;
          #pragma unroll
          for (int i = 0; i < 8; ++i) d[i] = (bf16_t)(acc[i] + bias);
        }
      }
    }
  }
  __syncthreads();

  // ---- Phase 3: attention for head w (S = QK^T, softmax, O = P V) ----
  const bf16_t* qh = sQ  + (size_t)w * NPAD * HD;
  const bf16_t* kh = sK  + (size_t)w * NPAD * HD;
  const bf16_t* vh = sVt + (size_t)w * HD * NPAD;
  bf16_t*       ph = sP  + (size_t)w * NPAD * NPAD;
  const float*  bh = sB  + (size_t)w * NTOK * NTOK;

  for (int mt = 0; mt < 4; ++mt) {
    v8f s[4];
    #pragma unroll
    for (int nt = 0; nt < 4; ++nt) {
      v16bf a  = ldA(qh, HD, mt * 16, 0, lane);
      v16bf bb = ldB(kh, HD, nt * 16, 0, lane);   // B = K^T, col = key index
      v8f z = {};
      s[nt] = wmma_bf16(a, bb, z);
    }
    // bias + mask + key padding (cols >= 49 -> -inf)
    #pragma unroll
    for (int nt = 0; nt < 4; ++nt) {
      int n = nt * 16 + lr;
      #pragma unroll
      for (int i = 0; i < 8; ++i) {
        int m = mt * 16 + half * 8 + i;
        float sv = s[nt][i];
        if (n >= NTOK)      sv = -1e30f;
        else if (m < NTOK)  sv += bh[m * NTOK + n];
        s[nt][i] = sv;
      }
    }
    // row softmax: rows live in 16-lane halves; xor 1/2/4/8 stays in-half
    #pragma unroll
    for (int i = 0; i < 8; ++i) {
      float mx = fmaxf(fmaxf(s[0][i], s[1][i]), fmaxf(s[2][i], s[3][i]));
      mx = fmaxf(mx, __shfl_xor(mx, 1));
      mx = fmaxf(mx, __shfl_xor(mx, 2));
      mx = fmaxf(mx, __shfl_xor(mx, 4));
      mx = fmaxf(mx, __shfl_xor(mx, 8));
      float e0 = __expf(s[0][i] - mx);
      float e1 = __expf(s[1][i] - mx);
      float e2 = __expf(s[2][i] - mx);
      float e3 = __expf(s[3][i] - mx);
      float sm = e0 + e1 + e2 + e3;
      sm += __shfl_xor(sm, 1);
      sm += __shfl_xor(sm, 2);
      sm += __shfl_xor(sm, 4);
      sm += __shfl_xor(sm, 8);
      float inv = 1.0f / sm;
      s[0][i] = e0 * inv; s[1][i] = e1 * inv;
      s[2][i] = e2 * inv; s[3][i] = e3 * inv;
    }
    // write P (bf16) rows for this mt; same-wave LDS ops are in-order
    #pragma unroll
    for (int nt = 0; nt < 4; ++nt) {
      bf16_t* d = ph + (size_t)(mt * 16 + half * 8) * NPAD + nt * 16 + lr;
      #pragma unroll
      for (int i = 0; i < 8; ++i) d[i * NPAD] = (bf16_t)(s[nt][i]);
    }
    // O tile = P[mt rows] @ V  (K = 64 keys = 2 WMMA steps)
    #pragma unroll
    for (int nt2 = 0; nt2 < 2; ++nt2) {
      v8f acc = {};
      #pragma unroll
      for (int kt = 0; kt < 2; ++kt) {
        v16bf a  = ldA(ph, NPAD, mt * 16, kt * 32, lane);
        v16bf bb = ldB(vh, NPAD, nt2 * 16, kt * 32, lane);
        acc = wmma_bf16(a, bb, acc);
      }
      bf16_t* d = sX + (size_t)(mt * 16 + half * 8) * DIM + w * HD + nt2 * 16 + lr;
      #pragma unroll
      for (int i = 0; i < 8; ++i) d[i * DIM] = (bf16_t)acc[i];
    }
  }
  __syncthreads();

  // ---- Phase 4: output projection, head w owns 32 output columns ----
  {
    float* ob = out + (size_t)b * NTOK * DIM;
    for (int mt = 0; mt < 4; ++mt) {
      for (int nt = 0; nt < 2; ++nt) {
        v8f acc = {};
        #pragma unroll
        for (int kt = 0; kt < 8; ++kt) {
          v16bf a  = ldA(sX, DIM, mt * 16, kt * 32, lane);
          v16bf bb = ldB(wproj, DIM, w * HD + nt * 16, kt * 32, lane);
          acc = wmma_bf16(a, bb, acc);
        }
        int   col = w * HD + nt * 16 + lr;
        float pb  = proj_b[col];
        int   rowb = mt * 16 + half * 8;
        #pragma unroll
        for (int i = 0; i < 8; ++i) {
          int row = rowb + i;
          if (row < NTOK) ob[(size_t)row * DIM + col] = acc[i] + pb;
        }
      }
    }
  }
}

// ---------------- launcher ----------------

extern "C" void kernel_launch(void* const* d_in, const int* in_sizes, int n_in,
                              void* d_out, int out_size, void* d_ws, size_t ws_size,
                              hipStream_t stream) {
  (void)in_sizes; (void)n_in; (void)out_size; (void)ws_size;
  const float* x          = (const float*)d_in[0];
  const float* mask       = (const float*)d_in[1];
  const float* qkv_w      = (const float*)d_in[2];
  const float* qkv_b      = (const float*)d_in[3];
  const float* proj_w     = (const float*)d_in[4];
  const float* proj_b     = (const float*)d_in[5];
  const float* bias_table = (const float*)d_in[6];
  const int*   rel_index  = (const int*)d_in[7];
  float* out = (float*)d_out;

  bf16_t* wq = (bf16_t*)d_ws;                 // 768*256 bf16
  bf16_t* wp = wq + QKV_ELEMS;                // 256*256 bf16

  convert_weights_kernel<<<(QKV_ELEMS + 255) / 256, 256, 0, stream>>>(
      qkv_w, proj_w, wq, wp);

  const size_t smem_bytes = 207936 + (size_t)HEADS * NPAD * NPAD * 2;  // 273472
  static bool attr_set = false;
  if (!attr_set) {
    hipFuncSetAttribute((const void*)win_attn_kernel,
                        hipFuncAttributeMaxDynamicSharedMemorySize,
                        (int)smem_bytes);
    attr_set = true;
  }
  win_attn_kernel<<<4096, 256, smem_bytes, stream>>>(
      x, mask, qkv_b, proj_b, bias_table, rel_index, wq, wp, out);
}
